// DynamicRouter_56959856280360
// MI455X (gfx1250) — compile-verified, matrix operand order checked by reading, not run on hardware
//
#include <hip/hip_runtime.h>
#include <stdint.h>
#include <stddef.h>

// ---------------- types ----------------
typedef __attribute__((ext_vector_type(16))) __bf16 v16bf;
typedef __attribute__((ext_vector_type(8)))  float  v8f;
typedef __attribute__((ext_vector_type(4)))  float  f4;

// ---------------- problem dims ----------------
#define B_ROWS 8192
#define DDIM   4096
#define EEXP   64
#define TOPK   2

#define KCHUNK 32                       // K per bf16 WMMA
#define NCHUNK (DDIM / KCHUNK)          // 128 chunks
#define NTILES (EEXP / 16)              // 4 N-tiles of 16
#define FRAG_HALF 16                    // bf16 elems per lane per fragment
#define CHUNK_ELEMS (NTILES * 32 * FRAG_HALF)  // 2048 bf16 = 4KB per chunk

#define WAVES 4                         // waves per block (wave32)
#define BLOCK_M (WAVES * 16)            // 64 rows per block

#define WS_W_BYTES (NCHUNK * CHUNK_ELEMS * 2)  // 512 KB swizzled bf16 W

// =====================================================================
// Kernel 1: convert gate_w [E,D] fp32 -> bf16, pre-swizzled into WMMA
// B-fragment order (ISA 7.12.2 16-bit layout): fragment(c,t) is 32 lanes
// x 16 bf16 contiguous, so the GEMM kernel's LDS reads are conflict-free
// b128 loads.  Lane l (group g=l>>4) holds B[k][n] with n = t*16+(l&15),
// k(chunk-relative) = {8g+0..7, 16+8g+0..7}.
// =====================================================================
__global__ __launch_bounds__(256) void router_prep_w(
    const float* __restrict__ W, __bf16* __restrict__ wswz)
{
    int tid = blockIdx.x * blockDim.x + threadIdx.x;   // 0..16383
    int l = tid & 31;
    int F = tid >> 5;                                  // fragment id: c*4+t
    int t = F & 3;
    int c = F >> 2;
    int n = t * 16 + (l & 15);
    int g = l >> 4;
    const float* src = W + (size_t)n * DDIM + c * KCHUNK + 8 * g;
    __bf16* dst = wswz + (size_t)F * (32 * FRAG_HALF) + l * FRAG_HALF;
#pragma unroll
    for (int i = 0; i < 8; ++i) {
        dst[i]     = (__bf16)src[i];        // k = 8g + i
        dst[8 + i] = (__bf16)src[16 + i];   // k = 16 + 8g + i
    }
}

// =====================================================================
// Kernel 2: logits = x @ W^T  via v_wmma_f32_16x16x32_bf16.
// 4 waves/block; wave w owns rows [blk*64 + w*16, +16) x all 64 experts.
// W chunks double-buffered in LDS via async global->LDS copies.
// =====================================================================
__global__ __launch_bounds__(WAVES * 32) void router_gemm(
    const float* __restrict__ x, const __bf16* __restrict__ wswz,
    float* __restrict__ logits)
{
    __shared__ __align__(64) __bf16 lds[2 * CHUNK_ELEMS];   // 8 KB

    const int tid  = threadIdx.x;
    const int wave = tid >> 5;
    const int lane = tid & 31;
    const int m    = lane & 15;    // row within tile
    const int g    = lane >> 4;    // lane group

    const int rowBase = blockIdx.x * BLOCK_M + wave * 16;
    const float* xrow = x + (size_t)(rowBase + m) * DDIM + 8 * g;

    // each of the 128 threads stages 32B (2 x b128) of the 4KB chunk
    const uint32_t lbase0 = (uint32_t)(uintptr_t)(&lds[0]) + tid * 32u;
    const uint32_t lbase1 = lbase0 + CHUNK_ELEMS * 2u;

    v8f acc[NTILES];
#pragma unroll
    for (int t = 0; t < NTILES; ++t)
        acc[t] = (v8f){0.f, 0.f, 0.f, 0.f, 0.f, 0.f, 0.f, 0.f};

    auto issue_chunk = [&](int c) {
        uint64_t ga = (uint64_t)(uintptr_t)(wswz + (size_t)c * CHUNK_ELEMS) + tid * 32u;
        uint32_t la = (c & 1) ? lbase1 : lbase0;
        asm volatile("global_load_async_to_lds_b128 %0, %1, off"
                     :: "v"(la), "v"(ga) : "memory");
        asm volatile("global_load_async_to_lds_b128 %0, %1, off"
                     :: "v"(la + 16u), "v"(ga + 16u) : "memory");
    };

    issue_chunk(0);

    for (int c = 0; c < NCHUNK; ++c) {
        if (c + 1 < NCHUNK) {
            issue_chunk(c + 1);
            asm volatile("s_wait_asynccnt 0x2" ::: "memory"); // chunk c done
        } else {
            asm volatile("s_wait_asynccnt 0x0" ::: "memory");
        }
        __syncthreads();   // publish LDS writes across waves

        // ---- A fragment: 16 fp32 -> 16 bf16, ISA 16-bit A 16x32 layout
        const float* xp = xrow + c * KCHUNK;
        f4 a0 = *(const f4*)(xp);        // k = 8g + 0..3
        f4 a1 = *(const f4*)(xp + 4);    // k = 8g + 4..7
        f4 a2 = *(const f4*)(xp + 16);   // k = 16 + 8g + 0..3
        f4 a3 = *(const f4*)(xp + 20);   // k = 16 + 8g + 4..7
        v16bf afrag;
#pragma unroll
        for (int i = 0; i < 4; ++i) {
            afrag[i]      = (__bf16)a0[i];
            afrag[4 + i]  = (__bf16)a1[i];
            afrag[8 + i]  = (__bf16)a2[i];
            afrag[12 + i] = (__bf16)a3[i];
        }

        // ---- 2-stage pipeline over the 4 N-tiles: only two B fragments
        // live at once (low pressure -> in-place C/D accumulation), while
        // the ds_load of tile t+1 overlaps the WMMA of tile t.
        const __bf16* lbuf = &lds[(c & 1) * CHUNK_ELEMS];
        v16bf bA = *(const v16bf*)(lbuf + 0 * (32 * FRAG_HALF) + lane * FRAG_HALF);
        v16bf bB = *(const v16bf*)(lbuf + 1 * (32 * FRAG_HALF) + lane * FRAG_HALF);

        acc[0] = __builtin_amdgcn_wmma_f32_16x16x32_bf16(
            false, afrag, false, bA, (short)0, acc[0], false, false);
        bA = *(const v16bf*)(lbuf + 2 * (32 * FRAG_HALF) + lane * FRAG_HALF);
        acc[1] = __builtin_amdgcn_wmma_f32_16x16x32_bf16(
            false, afrag, false, bB, (short)0, acc[1], true, false);
        bB = *(const v16bf*)(lbuf + 3 * (32 * FRAG_HALF) + lane * FRAG_HALF);
        acc[2] = __builtin_amdgcn_wmma_f32_16x16x32_bf16(
            false, afrag, false, bA, (short)0, acc[2], true, false);
        acc[3] = __builtin_amdgcn_wmma_f32_16x16x32_bf16(
            false, afrag, false, bB, (short)0, acc[3], true, false);

        __syncthreads();   // buffer (c&1) free for reuse by chunk c+2
    }

    // ---- epilogue: C layout (VGPR r, lane l) -> M = r + 8g, N = t*16 + m
#pragma unroll
    for (int t = 0; t < NTILES; ++t)
#pragma unroll
        for (int r = 0; r < 8; ++r)
            logits[(size_t)(rowBase + r + 8 * g) * EEXP + t * 16 + m] = acc[t][r];
}

// =====================================================================
// Kernel 3: per-row top-2 + softmax + scatter.  One thread per row.
// =====================================================================
__global__ __launch_bounds__(256) void router_topk(
    const float* __restrict__ logits, const float* __restrict__ temperature,
    float* __restrict__ rm, int* __restrict__ top_idx)
{
    int row = blockIdx.x * blockDim.x + threadIdx.x;
    if (row >= B_ROWS) return;
    const float invT = 1.0f / temperature[0];
    const float* lr = logits + (size_t)row * EEXP;

    float b1 = -3.4e38f, b2 = -3.4e38f;
    int i1 = 0, i2 = 0;
#pragma unroll 4
    for (int e = 0; e < EEXP; ++e) {
        float v = lr[e] * invT;
        if (v > b1)      { b2 = b1; i2 = i1; b1 = v; i1 = e; }
        else if (v > b2) { b2 = v; i2 = e; }
    }
    float w2 = __expf(b2 - b1);         // softmax over {b1, b2}, max-shifted
    float inv = 1.0f / (1.0f + w2);

    float* out = rm + (size_t)row * EEXP;
#pragma unroll 4
    for (int e = 0; e < EEXP; ++e) out[e] = 0.0f;
    out[i1] = inv;
    out[i2] = w2 * inv;
    top_idx[row * TOPK]     = i1;
    top_idx[row * TOPK + 1] = i2;
}

// =====================================================================
extern "C" void kernel_launch(void* const* d_in, const int* in_sizes, int n_in,
                              void* d_out, int out_size, void* d_ws, size_t ws_size,
                              hipStream_t stream)
{
    const float* x    = (const float*)d_in[0];   // [B, D] fp32
    const float* gw   = (const float*)d_in[1];   // [E, D] fp32
    const float* temp = (const float*)d_in[2];   // scalar

    __bf16* wswz  = (__bf16*)d_ws;                               // 512 KB
    float* logits = (float*)((char*)d_ws + WS_W_BYTES);          // 2 MB

    float* rm    = (float*)d_out;                                // [B, E]
    int*   tidx  = (int*)(rm + (size_t)B_ROWS * EEXP);           // [B, K]

    router_prep_w<<<(NCHUNK * NTILES * 32) / 256, 256, 0, stream>>>(gw, wswz);
    router_gemm<<<B_ROWS / BLOCK_M, WAVES * 32, 0, stream>>>(x, wswz, logits);
    router_topk<<<B_ROWS / 256, 256, 0, stream>>>(logits, temp, rm, tidx);
}
